// FlashAttention2_46677704573303
// MI455X (gfx1250) — matrix operand order repaired and checked by reading, not compiled, not run on hardware
//
#include <hip/hip_runtime.h>

// ---------------------------------------------------------------------------
// Problem constants (from the reference)
// ---------------------------------------------------------------------------
constexpr int kB   = 2;
constexpr int kS   = 2048;
constexpr int kH   = 4096;
constexpr int kNH  = 32;
constexpr int kNKV = 8;
constexpr int kHD  = 128;
constexpr int kKVD = kNKV * kHD;   // 1024
constexpr int kM   = kB * kS;      // 4096 rows of the token matrix

typedef __attribute__((ext_vector_type(16))) __bf16 v16bf;
typedef __attribute__((ext_vector_type(8)))  float  v8f;

#define WMMA_BF16(a, b, c) \
  __builtin_amdgcn_wmma_f32_16x16x32_bf16(false, (a), false, (b), (short)0, (c), false, false)

__device__ inline v8f zero8() {
  v8f z = {0.f, 0.f, 0.f, 0.f, 0.f, 0.f, 0.f, 0.f};
  return z;
}

// ---------------------------------------------------------------------------
// CDNA5 async global->LDS copy (ASYNCcnt-tracked DMA-like path).
// Per ISA 10.2 the low 32 bits of a generic pointer to __shared__ are the LDS
// byte address, which is what the instruction's VDST operand expects.
// ---------------------------------------------------------------------------
__device__ inline void async_copy_b128(void* lds_dst, const void* gsrc) {
  unsigned lds_off = (unsigned)(size_t)lds_dst;
  asm volatile("global_load_async_to_lds_b128 %0, %1, off"
               :: "v"(lds_off), "v"((unsigned long long)(size_t)gsrc)
               : "memory");
}
__device__ inline void wait_async() {
  asm volatile("s_wait_asynccnt 0" ::: "memory");
}

// ---------------------------------------------------------------------------
// fp32 -> bf16 conversion (memory-bound, trivial cost)
// ---------------------------------------------------------------------------
__global__ void cvt_f32_bf16(const float* __restrict__ in, __bf16* __restrict__ out, int n) {
  int i = (blockIdx.x * 256 + threadIdx.x) * 4;
  if (i >= n) return;
  float4 v = *(const float4*)(in + i);
  out[i + 0] = (__bf16)v.x;
  out[i + 1] = (__bf16)v.y;
  out[i + 2] = (__bf16)v.z;
  out[i + 3] = (__bf16)v.w;
}

// ---------------------------------------------------------------------------
// In-place RoPE on bf16 [kM][heads*128]; fp32 trig
// ---------------------------------------------------------------------------
__global__ void rope_kernel(__bf16* x, int heads, int total) {
  int idx = blockIdx.x * 256 + threadIdx.x;
  if (idx >= total) return;
  int j    = idx & 63;         // rotation pair index within head-dim/2
  int rest = idx >> 6;
  int hh   = rest % heads;
  int row  = rest / heads;     // 0..kM-1
  int s    = row & (kS - 1);   // position id
  __bf16* p = x + (size_t)row * heads * kHD + hh * kHD;
  // theta^(-j/64) = exp2(-j * log2(10000)/64)
  float inv = exp2f(-(float)j * 0.2076204562f);
  float ang = (float)s * inv;
  float cs = cosf(ang), sn = sinf(ang);
  float x1 = (float)p[j], x2 = (float)p[j + 64];
  p[j]      = (__bf16)(x1 * cs - x2 * sn);
  p[j + 64] = (__bf16)(x2 * cs + x1 * sn);
}

// ---------------------------------------------------------------------------
// bf16 WMMA GEMM: C[M,N] = A[M,K] * B[K,N]
// BM=128 BN=256 BK=32, 256 threads = 8 waves (2x4), each wave 64x64 (4x4 WMMA)
// A tile staged via GLOBAL_LOAD_ASYNC_TO_LDS_B128 (double buffered, overlapped
// with the 16-WMMA compute); B staged transposed + pair-packed in VGPRs.
// ---------------------------------------------------------------------------
template <typename OutT>
__global__ __launch_bounds__(256) void gemm_bf16(const __bf16* __restrict__ A,
                                                 const __bf16* __restrict__ Bw,
                                                 OutT* __restrict__ C,
                                                 int M, int N, int K) {
  constexpr int BM = 128, BN = 256, BK = 32;
  __shared__ __bf16   As[2][BM][BK + 8];          // row pitch 80B = 5x16B
  __shared__ unsigned BsT[2][BN][BK / 2 + 4];     // [n][k-pair] packed bf16 pairs

  const int t    = threadIdx.x;
  const int lane = t & 31, w = t >> 5;
  const int wm   = w >> 2, wn = w & 3;            // 2x4 wave grid
  const int lo   = lane & 15, hi = (lane >> 4) & 1;
  const int bm   = blockIdx.y * BM, bn = blockIdx.x * BN;

  v8f acc[4][4];
#pragma unroll
  for (int i = 0; i < 4; ++i)
#pragma unroll
    for (int j = 0; j < 4; ++j) acc[i][j] = zero8();

  const int arow = t >> 1, acb = (t & 1) * 16;    // A stage: 2 thr/row, 16 elems each

  auto stageA = [&](int buf, int ko) {            // async DMA, no VGPR round-trip
    const __bf16* ap = A + (size_t)(bm + arow) * K + ko * BK + acb;
    async_copy_b128(&As[buf][arow][acb], ap);
    async_copy_b128(&As[buf][arow][acb + 8], ap + 8);
  };

  uint4 rbv[2][2];
  auto gloadB = [&](int ko) {
#pragma unroll
    for (int it = 0; it < 2; ++it) {
      const int task = t + it * 256;
      const int kpr = task >> 5, n0 = (task & 31) * 8;
      const __bf16* bp = Bw + (size_t)(ko * BK + 2 * kpr) * N + bn + n0;
      rbv[it][0] = *(const uint4*)bp;
      rbv[it][1] = *(const uint4*)(bp + N);
    }
  };
  auto sstoreB = [&](int buf) {
#pragma unroll
    for (int it = 0; it < 2; ++it) {
      const int task = t + it * 256;
      const int kpr = task >> 5, n0 = (task & 31) * 8;
      unsigned r0[4] = {rbv[it][0].x, rbv[it][0].y, rbv[it][0].z, rbv[it][0].w};
      unsigned r1[4] = {rbv[it][1].x, rbv[it][1].y, rbv[it][1].z, rbv[it][1].w};
#pragma unroll
      for (int j = 0; j < 4; ++j) {
        BsT[buf][n0 + 2 * j][kpr]     = (r0[j] & 0xffffu) | ((r1[j] & 0xffffu) << 16);
        BsT[buf][n0 + 2 * j + 1][kpr] = (r0[j] >> 16)     | ((r1[j] >> 16) << 16);
      }
    }
  };

  const int nk = K / BK;
  stageA(0, 0);
  gloadB(0);
  sstoreB(0);
  wait_async();
  __syncthreads();

  for (int ko = 0; ko < nk; ++ko) {
    const int cur = ko & 1;
    if (ko + 1 < nk) {
      stageA(cur ^ 1, ko + 1);                    // async into other buffer
      gloadB(ko + 1);
    }
    if (ko + 2 < nk) {
      // keep GL2 ahead of the LDS double-buffer (global_prefetch_b8)
      __builtin_prefetch(Bw + (size_t)((ko + 2) * BK + 2 * (t >> 5)) * N + bn + (t & 31) * 8, 0, 1);
    }

    v16bf af[4], bfr[4];
#pragma unroll
    for (int mt = 0; mt < 4; ++mt) {
      const __bf16* p = &As[cur][wm * 64 + mt * 16 + lo][0];
      union { v16bf v; unsigned u[8]; } f;
#pragma unroll
      for (int vv = 0; vv < 8; ++vv) {
        int k = (hi << 4) + (vv << 1);   // A frag layout
        f.u[vv] = *(const unsigned*)(p + k);
      }
      af[mt] = f.v;
    }
#pragma unroll
    for (int nt = 0; nt < 4; ++nt) {
      const unsigned* p = &BsT[cur][wn * 64 + nt * 16 + lo][hi * 8]; // B frag layout
      union { v16bf v; unsigned u[8]; } f;
#pragma unroll
      for (int vv = 0; vv < 8; ++vv) f.u[vv] = p[vv];
      bfr[nt] = f.v;
    }
#pragma unroll
    for (int mt = 0; mt < 4; ++mt)
#pragma unroll
      for (int nt = 0; nt < 4; ++nt)
        acc[mt][nt] = WMMA_BF16(af[mt], bfr[nt], acc[mt][nt]);

    if (ko + 1 < nk) sstoreB(cur ^ 1);
    wait_async();                                  // this wave's async A-stage done
    __syncthreads();
  }

#pragma unroll
  for (int mt = 0; mt < 4; ++mt)
#pragma unroll
    for (int nt = 0; nt < 4; ++nt) {
      int col = bn + wn * 64 + nt * 16 + lo;
#pragma unroll
      for (int r = 0; r < 8; ++r) {
        int row = bm + wm * 64 + mt * 16 + r + hi * 8;   // C/D VGPR layout
        C[(size_t)row * N + col] = (OutT)acc[mt][nt][r];
      }
    }
}

// ---------------------------------------------------------------------------
// Causal GQA flash attention, bf16 WMMA, fp32 online softmax.
// Block = 256 thr (8 waves), 128 q-rows per block (16 per wave), 64-kv chunks.
// K chunk staged with async global->LDS; V staged transposed via VGPR packing.
// ---------------------------------------------------------------------------
__global__ __launch_bounds__(256) void flash_attn(const __bf16* __restrict__ Q,
                                                  const __bf16* __restrict__ Kc,
                                                  const __bf16* __restrict__ Vc,
                                                  __bf16* __restrict__ O) {
  constexpr int BQ = 128, BKV = 64;
  __shared__ __bf16   Qt[BQ][kHD + 8];           // row pitch 272B = 17x16B
  __shared__ __bf16   Kt[BKV][kHD + 8];
  __shared__ unsigned VtT[kHD][BKV / 2 + 4];     // V transposed, pair-packed
  __shared__ __bf16   Pl[8][16][BKV + 8];        // per-wave P tile

  const int t = threadIdx.x, lane = t & 31, w = t >> 5;
  const int lo = lane & 15, hi = (lane >> 4) & 1;
  const int qt = blockIdx.x, h = blockIdx.y, b = blockIdx.z;
  const int kh = h >> 2;                         // GQA: NH/NKV = 4
  const int qrow0 = qt * BQ + w * 16;

  // stage Q (once, async)
  {
    const int row = t >> 1, cb = (t & 1) * 64;
    const __bf16* src = Q + (size_t)(b * kS + qt * BQ + row) * kH + h * kHD + cb;
#pragma unroll
    for (int j = 0; j < 8; ++j)
      async_copy_b128(&Qt[row][cb + j * 8], src + j * 8);
  }

  v8f o[8];
#pragma unroll
  for (int j = 0; j < 8; ++j) o[j] = zero8();
  float m_i[8], l_i[8];
#pragma unroll
  for (int r = 0; r < 8; ++r) { m_i[r] = -3.0e38f; l_i[r] = 0.0f; }

  const int nchunks = 2 * (qt + 1);
  for (int jc = 0; jc < nchunks; ++jc) {
    const int kbase = jc * BKV;
    // stage K chunk (row-major, async)
    {
      const int row = t >> 2, cb = (t & 3) * 32;
      const __bf16* src = Kc + (size_t)(b * kS + kbase + row) * kKVD + kh * kHD + cb;
#pragma unroll
      for (int j = 0; j < 4; ++j)
        async_copy_b128(&Kt[row][cb + j * 8], src + j * 8);
    }
    // stage V^T (pair-packed along kv)
#pragma unroll
    for (int it = 0; it < 2; ++it) {
      const int task = t + it * 256;
      const int kvp = task >> 4, d0 = (task & 15) * 8;
      const __bf16* src = Vc + (size_t)(b * kS + kbase + kvp * 2) * kKVD + kh * kHD + d0;
      uint4 r0 = *(const uint4*)src;
      uint4 r1 = *(const uint4*)(src + kKVD);
      unsigned a0[4] = {r0.x, r0.y, r0.z, r0.w};
      unsigned a1[4] = {r1.x, r1.y, r1.z, r1.w};
#pragma unroll
      for (int j = 0; j < 4; ++j) {
        VtT[d0 + 2 * j][kvp]     = (a0[j] & 0xffffu) | ((a1[j] & 0xffffu) << 16);
        VtT[d0 + 2 * j + 1][kvp] = (a0[j] >> 16)     | ((a1[j] >> 16) << 16);
      }
      if (jc + 1 < nchunks)
        __builtin_prefetch(src + (size_t)BKV * kKVD, 0, 1);
    }
    wait_async();
    __syncthreads();

    if (kbase <= qrow0 + 15) {   // chunk has unmasked elements for this wave
      // ---- scores: S = Q * K^T (16 WMMAs) ----
      v8f sc[4] = {zero8(), zero8(), zero8(), zero8()};
#pragma unroll
      for (int ks = 0; ks < 4; ++ks) {
        union { v16bf v; unsigned u[8]; } fa;
        const __bf16* qp = &Qt[w * 16 + lo][ks * 32];
#pragma unroll
        for (int vv = 0; vv < 8; ++vv) {
          int k = (hi << 4) + (vv << 1);
          fa.u[vv] = *(const unsigned*)(qp + k);
        }
#pragma unroll
        for (int nt = 0; nt < 4; ++nt) {
          union { v16bf v; unsigned u[8]; } fb;
          const __bf16* kp = &Kt[nt * 16 + lo][ks * 32 + (hi << 4)];
#pragma unroll
          for (int vv = 0; vv < 8; ++vv) fb.u[vv] = *(const unsigned*)(kp + 2 * vv);
          sc[nt] = WMMA_BF16(fa.v, fb.v, sc[nt]);
        }
      }
      // ---- online softmax (rows spread over 16 lanes) ----
      const float scale = 0.08838834764831845f;  // 1/sqrt(128)
      float corr[8];
#pragma unroll
      for (int r = 0; r < 8; ++r) {
        float mx = -3.0e38f;
#pragma unroll
        for (int nt = 0; nt < 4; ++nt) {
          float s = (float)sc[nt][r] * scale;
          int kvg = kbase + nt * 16 + lo;
          int qg  = qrow0 + r + 8 * hi;
          s = (kvg <= qg) ? s : -3.0e38f;       // causal mask
          sc[nt][r] = s;
          mx = fmaxf(mx, s);
        }
#pragma unroll
        for (int msk = 1; msk < 16; msk <<= 1) mx = fmaxf(mx, __shfl_xor(mx, msk, 32));
        float mn = fmaxf(m_i[r], mx);
        float c  = __expf(m_i[r] - mn);
        float rs = 0.0f;
#pragma unroll
        for (int nt = 0; nt < 4; ++nt) {
          float p = __expf((float)sc[nt][r] - mn);
          sc[nt][r] = p;
          rs += p;
        }
#pragma unroll
        for (int msk = 1; msk < 16; msk <<= 1) rs += __shfl_xor(rs, msk, 32);
        m_i[r]  = mn;
        l_i[r]  = l_i[r] * c + rs;
        corr[r] = c;
      }
#pragma unroll
      for (int nt = 0; nt < 8; ++nt)
#pragma unroll
        for (int r = 0; r < 8; ++r) o[nt][r] = o[nt][r] * corr[r];
      // ---- P to LDS (per-wave region, A-fragment layout friendly) ----
#pragma unroll
      for (int nt = 0; nt < 4; ++nt)
#pragma unroll
        for (int r = 0; r < 8; ++r)
          Pl[w][r + 8 * hi][nt * 16 + lo] = (__bf16)(float)sc[nt][r];
      // ---- O += P * V (16 WMMAs) ----
#pragma unroll
      for (int ks = 0; ks < 2; ++ks) {
        union { v16bf v; unsigned u[8]; } fa;
        const __bf16* pp = &Pl[w][lo][ks * 32];
#pragma unroll
        for (int vv = 0; vv < 8; ++vv) {
          int k = (hi << 4) + (vv << 1);
          fa.u[vv] = *(const unsigned*)(pp + k);
        }
#pragma unroll
        for (int nt = 0; nt < 8; ++nt) {
          union { v16bf v; unsigned u[8]; } fb;
          const unsigned* vp = &VtT[nt * 16 + lo][ks * 16 + hi * 8];
#pragma unroll
          for (int vv = 0; vv < 8; ++vv) fb.u[vv] = vp[vv];
          o[nt] = WMMA_BF16(fa.v, fb.v, o[nt]);
        }
      }
    }
    __syncthreads();
  }

#pragma unroll
  for (int r = 0; r < 8; ++r) l_i[r] = 1.0f / l_i[r];
#pragma unroll
  for (int nt = 0; nt < 8; ++nt) {
    int col = h * kHD + nt * 16 + lo;
#pragma unroll
    for (int r = 0; r < 8; ++r) {
      int row = b * kS + qrow0 + r + 8 * hi;
      O[(size_t)row * kH + col] = (__bf16)(o[nt][r] * l_i[r]);
    }
  }
}

// ---------------------------------------------------------------------------
// Host launcher
// ---------------------------------------------------------------------------
extern "C" void kernel_launch(void* const* d_in, const int* in_sizes, int n_in,
                              void* d_out, int out_size, void* d_ws, size_t ws_size,
                              hipStream_t stream) {
  (void)in_sizes; (void)n_in; (void)out_size; (void)ws_size;
  const float* hs = (const float*)d_in[0];
  const float* Wq = (const float*)d_in[1];
  const float* Wk = (const float*)d_in[2];
  const float* Wv = (const float*)d_in[3];
  const float* Wo = (const float*)d_in[4];
  float* out = (float*)d_out;

  char* ws = (char*)d_ws;
  size_t off = 0;
  auto take = [&](size_t elems) {
    void* p = ws + off;
    off += (elems * 2 + 255) & ~(size_t)255;
    return (__bf16*)p;
  };
  __bf16* Xb  = take((size_t)kM * kH);
  __bf16* Wqb = take((size_t)kH * kH);
  __bf16* Wkb = take((size_t)kH * kKVD);
  __bf16* Wvb = take((size_t)kH * kKVD);
  __bf16* Wob = take((size_t)kH * kH);
  __bf16* Qb  = take((size_t)kM * kH);
  __bf16* Kb  = take((size_t)kM * kKVD);
  __bf16* Vb  = take((size_t)kM * kKVD);
  __bf16* Ob  = take((size_t)kM * kH);

  auto cvt = [&](const float* src, __bf16* dst, size_t n) {
    cvt_f32_bf16<<<dim3((unsigned)((n / 4 + 255) / 256)), 256, 0, stream>>>(src, dst, (int)n);
  };
  cvt(hs, Xb, (size_t)kM * kH);
  cvt(Wq, Wqb, (size_t)kH * kH);
  cvt(Wk, Wkb, (size_t)kH * kKVD);
  cvt(Wv, Wvb, (size_t)kH * kKVD);
  cvt(Wo, Wob, (size_t)kH * kH);

  gemm_bf16<__bf16><<<dim3(kH / 256, kM / 128), 256, 0, stream>>>(Xb, Wqb, Qb, kM, kH, kH);
  gemm_bf16<__bf16><<<dim3(kKVD / 256, kM / 128), 256, 0, stream>>>(Xb, Wkb, Kb, kM, kKVD, kH);
  gemm_bf16<__bf16><<<dim3(kKVD / 256, kM / 128), 256, 0, stream>>>(Xb, Wvb, Vb, kM, kKVD, kH);

  rope_kernel<<<(kM * kNH * 64) / 256, 256, 0, stream>>>(Qb, kNH, kM * kNH * 64);
  rope_kernel<<<(kM * kNKV * 64) / 256, 256, 0, stream>>>(Kb, kNKV, kM * kNKV * 64);

  flash_attn<<<dim3(kS / 128, kNH, kB), 256, 0, stream>>>(Qb, Kb, Vb, Ob);

  gemm_bf16<float><<<dim3(kH / 256, kM / 128), 256, 0, stream>>>(Ob, Wob, out, kM, kH, kH);
}
